// AudiomerEncoderBlock_18270790877527
// MI455X (gfx1250) — matrix operand order repaired and checked by reading, not compiled, not run on hardware
//
#include <hip/hip_runtime.h>
#include <hip/hip_bf16.h>

// ---------------------------------------------------------------------------
// Audiomer encoder block for gfx1250 (MI455X).
//  - MBConv stage: LDS-tiled f32 VALU (bandwidth-shaped, exact)
//  - All attention / FF GEMMs: v_wmma_f32_16x16x32_bf16 (bf16 in, f32 accum)
//  - LDS panels stored in WMMA fragment order -> ds_load_b128 fragment reads
// ---------------------------------------------------------------------------

typedef __attribute__((ext_vector_type(16))) __bf16 v16bf;
typedef __attribute__((ext_vector_type(8)))  float  v8f;

#define BB    16
#define LL    8192
#define NQ    4096
#define NK    2048
#define CC    128
#define INNER 384
#define HH    6
#define DHD   64
#define MF    266
#define MP    272
#define BHN   (BB*HH)
#define FFD   512
#define FFH   256
#define MQ    (BB*NQ)   /* 65536 */
#define MK    (BB*NK)   /* 32768 */

#define DN_   0.35355339059327373f   /* 64^-0.25 */
#define RATIO_ 0.06131393394849658f  /* 266^-0.5 */
#define FEPS_ 1e-4f

enum { EPI_BF16 = 0, EPI_F32 = 1, EPI_F32_RES = 2, EPI_BF16_RS = 3 };

static __device__ __forceinline__ unsigned short f2bf(float f) {
  union { float f; unsigned u; } x; x.f = f;
  unsigned u = x.u;
  unsigned r = (u + 0x7FFFu + ((u >> 16) & 1u)) >> 16;
  return (unsigned short)r;
}
static __device__ __forceinline__ float bf2f(unsigned short h) {
  union { unsigned u; float f; } x; x.u = ((unsigned)h) << 16; return x.f;
}
static __device__ __forceinline__ float silu_(float x) { return x / (1.f + __expf(-x)); }
static __device__ __forceinline__ unsigned fenc(float f) {
  unsigned u = __float_as_uint(f);
  return (u & 0x80000000u) ? ~u : (u | 0x80000000u);
}
static __device__ __forceinline__ float fdec(unsigned u) {
  unsigned b = (u & 0x80000000u) ? (u & 0x7FFFFFFFu) : ~u;
  return __uint_as_float(b);
}

// load a 32-byte WMMA fragment from LDS as 2x b128
static __device__ __forceinline__ v16bf ld_frag(const unsigned short* p) {
  union { uint4 q[2]; v16bf v; } u;
  const uint4* s = (const uint4*)p;
  u.q[0] = s[0];
  u.q[1] = s[1];
  return u.v;
}

// A-panel fragment-order index: element (row r in tile-stack, k col c of chunk)
// frag layout: [m_tile][lane][e], lane = (r&15) + 16*h
static __device__ __forceinline__ int fragA_idx(int r, int c) {
  int mt = r >> 4, lr = r & 15, h, e;
  h = c >> 4; e = c & 15;
  return ((mt * 32 + lr + 16 * h) << 4) + e;
}
// B-panel fragment-order index: element (k row kk of chunk, col cc)
static __device__ __forceinline__ int fragB_idx(int kk, int cc) {
  int half = kk >> 4, e = kk & 15;
  int nt = cc >> 4, lr = cc & 15;
  return ((nt * 32 + lr + 16 * half) << 4) + e;
}

// ---------------------------------------------------------------------------
// MBConv stage
// ---------------------------------------------------------------------------

__global__ __launch_bounds__(256) void k_expand(const float* __restrict__ x,
    const float* __restrict__ wexp, const float* __restrict__ s1,
    const float* __restrict__ b1, float* __restrict__ e) {
  __shared__ float xs[64][65];
  __shared__ float wsm[128][64];
  int l0 = blockIdx.x * 64, b = blockIdx.y, tid = threadIdx.x;
  for (int i = 0; i < 16; i++) {
    int idx = tid + i * 256; int ci = idx >> 6, ll = idx & 63;
    xs[ci][ll] = x[((size_t)b * 64 + ci) * LL + l0 + ll];
  }
  for (int i = 0; i < 32; i++) {
    int idx = tid + i * 256; int co = idx >> 6, ci = idx & 63;
    wsm[co][ci] = wexp[co * 64 + ci];
  }
  __syncthreads();
  int ll = tid & 63, g = tid >> 6;
  for (int co = g; co < 128; co += 4) {
    float acc = 0.f;
    for (int ci = 0; ci < 64; ci++) acc += wsm[co][ci] * xs[ci][ll];
    float v = s1[co] * acc + b1[co];
    e[((size_t)b * 128 + co) * LL + l0 + ll] = silu_(v);
  }
}

__global__ void k_dwconv(const float* __restrict__ e, const float* __restrict__ wdw,
    const float* __restrict__ s2, const float* __restrict__ b2,
    float* __restrict__ h, int Lo, int stride) {
  int t = blockIdx.x * 256 + threadIdx.x;
  int c = blockIdx.y, b = blockIdx.z;
  if (t >= Lo) return;
  size_t base = ((size_t)b * 128 + c) * LL;
  float acc = 0.f;
#pragma unroll
  for (int j = 0; j < 5; j++) {
    int pos = stride * t - 2 + j;
    if (pos >= 0 && pos < LL) acc += e[base + pos] * wdw[c * 5 + j];
  }
  float v = s2[c] * acc + b2[c];
  h[((size_t)b * 128 + c) * Lo + t] = silu_(v);
}

__global__ void k_semean(const float* __restrict__ h, float* __restrict__ mean, int Lo) {
  __shared__ float red[256];
  int blk = blockIdx.x, tid = threadIdx.x;
  float s = 0.f;
  for (int i = tid; i < Lo; i += 256) s += h[(size_t)blk * Lo + i];
  red[tid] = s; __syncthreads();
  for (int o = 128; o > 0; o >>= 1) { if (tid < o) red[tid] += red[tid + o]; __syncthreads(); }
  if (!tid) mean[blk] = red[0] / (float)Lo;
}

__global__ void k_se(const float* __restrict__ mean, const float* __restrict__ w1,
    const float* __restrict__ b1, const float* __restrict__ w2,
    const float* __restrict__ b2, float* __restrict__ sesc) {
  __shared__ float ms[128];
  __shared__ float s1s[32];
  int b = blockIdx.x, tid = threadIdx.x;
  ms[tid] = mean[b * 128 + tid];
  __syncthreads();
  if (tid < 32) {
    float a = 0.f;
    for (int ci = 0; ci < 128; ci++) a += w1[tid * 128 + ci] * ms[ci];
    s1s[tid] = silu_(a + b1[tid]);
  }
  __syncthreads();
  float a = 0.f;
  for (int r = 0; r < 32; r++) a += w2[tid * 32 + r] * s1s[r];
  sesc[b * 128 + tid] = 1.f / (1.f + __expf(-(a + b2[tid])));
}

__global__ __launch_bounds__(256) void k_project(const float* __restrict__ h,
    const float* __restrict__ sesc, const float* __restrict__ wp,
    const float* __restrict__ s3, const float* __restrict__ b3,
    float* __restrict__ outT, int Lo) {
  __shared__ float hs[64][65];
  __shared__ float wpS[128][64];
  int t0 = blockIdx.x * 64, b = blockIdx.y, tid = threadIdx.x;
  int co = tid & 127, tg = tid >> 7;
  float acc[32];
  for (int i = 0; i < 32; i++) acc[i] = 0.f;
  for (int k0 = 0; k0 < 128; k0 += 64) {
    for (int i = 0; i < 16; i++) {
      int idx = tid + i * 256; int ci = idx >> 6, tt = idx & 63;
      hs[ci][tt] = h[((size_t)b * 128 + k0 + ci) * Lo + t0 + tt] * sesc[b * 128 + k0 + ci];
    }
    for (int i = 0; i < 32; i++) {
      int idx = tid + i * 256; int cc = idx >> 6, ci = idx & 63;
      wpS[cc][ci] = wp[cc * 128 + k0 + ci];
    }
    __syncthreads();
    for (int i = 0; i < 32; i++) {
      int tt = tg + 2 * i; float a = 0.f;
      for (int ci = 0; ci < 64; ci++) a += wpS[co][ci] * hs[ci][tt];
      acc[i] += a;
    }
    __syncthreads();
  }
  for (int i = 0; i < 32; i++) {
    int tt = tg + 2 * i;
    outT[((size_t)(b * Lo + t0 + tt)) * 128 + co] = s3[co] * acc[i] + b3[co];
  }
}

// ---------------------------------------------------------------------------
// small helpers
// ---------------------------------------------------------------------------

__global__ void k_cvt(const float* __restrict__ s, unsigned short* __restrict__ d, size_t n) {
  size_t i = (size_t)blockIdx.x * 256 + threadIdx.x;
  if (i < n) d[i] = f2bf(s[i]);
}

__global__ void k_scale_norm(const float* __restrict__ src, unsigned short* __restrict__ dst,
                             const float* __restrict__ g, int M) {
  int r = blockIdx.x * 256 + threadIdx.x;
  if (r >= M) return;
  size_t base = (size_t)r * 128;
  float s = 0.f;
  for (int i = 0; i < 128; i++) { float v = src[base + i]; s += v * v; }
  float n = fmaxf(sqrtf(s), 1e-5f);
  float sc = g[0] / n;
  for (int i = 0; i < 128; i++) dst[base + i] = f2bf(src[base + i] * sc);
}

// projT[d][m] = dn * proj[m][d], zero-padded to 272 features
__global__ void k_projT(const float* __restrict__ proj, unsigned short* __restrict__ pjt) {
  int i = blockIdx.x * 256 + threadIdx.x;
  if (i >= 64 * MP) return;
  int d = i / MP, m = i % MP;
  float v = (m < MF) ? proj[m * 64 + d] * DN_ : 0.f;
  pjt[d * MP + m] = f2bf(v);
}

__global__ void k_zero(unsigned* p, int n) {
  int i = blockIdx.x * 256 + threadIdx.x;
  if (i < n) p[i] = 0u;
}

// ---------------------------------------------------------------------------
// Generic bf16 WMMA GEMM: C[bat][m][n] = A[bat][m][k] * B[bat][k][n] (+epilogue)
// block = 256 thr = 8 waves; block tile 64x32 of C, one 16x16 WMMA tile/wave.
// LDS panels stored fragment-major -> each lane reads its fragment as 2x b128.
// ---------------------------------------------------------------------------

struct GP {
  long bA0, bA1, sAm, sAk;
  long bB0, bB1, sBk, sBn;
  long bC0, bC1, ldc;
  long bR0, bR1, ldr;
  long bS0, bS1;
  int M, N, K, nH;
};

template <int EPI>
__global__ __launch_bounds__(256) void k_gemm(const unsigned short* __restrict__ A,
    const unsigned short* __restrict__ Bm, void* __restrict__ Cp,
    const float* __restrict__ bias, const float* __restrict__ res,
    const float* __restrict__ rs, GP p) {
  __shared__ __align__(16) unsigned short As2[4 * 32 * 16]; // [m_tile][lane][e]
  __shared__ __align__(16) unsigned short Bs2[2 * 32 * 16]; // [n_tile][lane][e]
  int zb = blockIdx.z / p.nH, zh = blockIdx.z % p.nH;
  const unsigned short* Ab = A + zb * p.bA0 + zh * p.bA1;
  const unsigned short* Bb = Bm + zb * p.bB0 + zh * p.bB1;
  int m_blk = blockIdx.x * 64, n_blk = blockIdx.y * 32;
  int tid = threadIdx.x, lane = tid & 31, wv = tid >> 5;
  int mt = wv & 3, nt = wv >> 2;
  int half = lane >> 4, lr = lane & 15;
  v8f acc = {0.f, 0.f, 0.f, 0.f, 0.f, 0.f, 0.f, 0.f};

  for (int k0 = 0; k0 < p.K; k0 += 32) {
#pragma unroll
    for (int i = 0; i < 8; i++) {
      int idx = tid + i * 256; int r = idx >> 5, c = idx & 31;
      int mm = m_blk + r, kk = k0 + c;
      unsigned short v = (mm < p.M && kk < p.K)
                             ? Ab[(long)mm * p.sAm + (long)kk * p.sAk]
                             : (unsigned short)0;
      As2[fragA_idx(r, c)] = v;
    }
#pragma unroll
    for (int i = 0; i < 4; i++) {
      int idx = tid + i * 256; int r = idx >> 5, c = idx & 31;
      int kk = k0 + r, nn = n_blk + c;
      unsigned short v = (kk < p.K && nn < p.N)
                             ? Bb[(long)kk * p.sBk + (long)nn * p.sBn]
                             : (unsigned short)0;
      Bs2[fragB_idx(r, c)] = v;
    }
    if (k0 + 32 < p.K)  // global_prefetch_b8 of next A panel
      __builtin_prefetch((const void*)(Ab + (long)(m_blk + (tid & 63)) * p.sAm +
                                       (long)(k0 + 32) * p.sAk), 0, 1);
    __syncthreads();
    v16bf a = ld_frag(As2 + ((mt * 32 + lane) << 4));
    v16bf b = ld_frag(Bs2 + ((nt * 32 + lane) << 4));
    acc = __builtin_amdgcn_wmma_f32_16x16x32_bf16(false, a, false, b, (short)0, acc,
                                                  false, false);
    __syncthreads();
  }
#pragma unroll
  for (int r = 0; r < 8; r++) {
    int mm = m_blk + mt * 16 + r + 8 * half, nn = n_blk + nt * 16 + lr;
    if (mm < p.M && nn < p.N) {
      float v = acc[r];
      if (bias) v += bias[nn];
      if (EPI == EPI_F32_RES) v += res[zb * p.bR0 + zh * p.bR1 + (long)mm * p.ldr + nn];
      if (EPI == EPI_BF16_RS) v *= rs[zb * p.bS0 + zh * p.bS1 + mm];
      long co = zb * p.bC0 + zh * p.bC1 + (long)mm * p.ldc + nn;
      if (EPI == EPI_F32 || EPI == EPI_F32_RES) ((float*)Cp)[co] = v;
      else ((unsigned short*)Cp)[co] = f2bf(v);
    }
  }
}

// ---------------------------------------------------------------------------
// FAVOR+ feature map kernels (fused WMMA GEMM + row stats + exp)
// Aq2 / Bp2 stored fragment-major; Aq row-major copy kept for diag.
// ---------------------------------------------------------------------------

__global__ __launch_bounds__(256) void k_qp(const unsigned short* __restrict__ qb,
    const unsigned short* __restrict__ pjt, unsigned short* __restrict__ qp) {
  __shared__ unsigned short Aq[16][66];
  __shared__ __align__(16) unsigned short Aq2[2 * 32 * 16];        // [chunk][lane][e]
  __shared__ __align__(16) unsigned short Bp2[17 * 2 * 32 * 16];   // [t][chunk][lane][e]
  __shared__ float XD[16][280];
  __shared__ float dg[16];
  __shared__ float rmx[16];
  int n0 = blockIdx.x * 16, bh = blockIdx.y;
  int b = bh / HH, hh = bh % HH;
  int tid = threadIdx.x;
  for (int i = 0; i < 4; i++) {
    int idx = tid + i * 256; int r = idx >> 6, d = idx & 63;
    unsigned short v = qb[((size_t)(b * NQ + n0 + r)) * INNER + hh * 64 + d];
    Aq[r][d] = v;
    Aq2[(d >> 5) * 512 + fragA_idx(r, d & 31)] = v;
  }
  for (int i = 0; i < 68; i++) {
    int idx = tid + i * 256;
    int d = idx / MP, m = idx % MP;
    Bp2[(((m >> 4) * 2 + (d >> 5)) << 9) + fragB_idx(d & 31, m & 15)] = pjt[idx];
  }
  __syncthreads();
  int lane = tid & 31, wv = tid >> 5, half = lane >> 4, lr = lane & 15;
  for (int t = wv; t < 17; t += 8) {
    v8f acc = {0.f, 0.f, 0.f, 0.f, 0.f, 0.f, 0.f, 0.f};
#pragma unroll
    for (int ch = 0; ch < 2; ch++) {
      v16bf a = ld_frag(Aq2 + ch * 512 + (lane << 4));
      v16bf bf = ld_frag(Bp2 + (((t * 2 + ch) * 32 + lane) << 4));
      acc = __builtin_amdgcn_wmma_f32_16x16x32_bf16(false, a, false, bf, (short)0, acc,
                                                    false, false);
    }
#pragma unroll
    for (int r = 0; r < 8; r++) XD[r + 8 * half][t * 16 + lr] = acc[r];
  }
  __syncthreads();
  if (tid < 16) {
    float s = 0.f;
    for (int d = 0; d < 64; d++) { float v = bf2f(Aq[tid][d]); s += v * v; }
    dg[tid] = s * (0.5f * DN_ * DN_);
    float m = -3.4e38f;
    for (int j = 0; j < MF; j++) m = fmaxf(m, XD[tid][j]);
    rmx[tid] = m;
  }
  __syncthreads();
  for (int i = tid; i < 16 * MP; i += 256) {
    int r = i / MP, m = i % MP;
    float v = (m < MF) ? RATIO_ * (__expf(XD[r][m] - dg[r] - rmx[r]) + FEPS_) : 0.f;
    qp[((size_t)bh * NQ + n0 + r) * MP + m] = f2bf(v);
  }
}

template <int PASS>
__global__ __launch_bounds__(256) void k_kxd(const unsigned short* __restrict__ kb,
    const unsigned short* __restrict__ pjt, unsigned* __restrict__ mxk,
    unsigned short* __restrict__ kp) {
  __shared__ unsigned short Aq[16][66];
  __shared__ __align__(16) unsigned short Aq2[2 * 32 * 16];
  __shared__ __align__(16) unsigned short Bp2[17 * 2 * 32 * 16];
  __shared__ float XD[16][280];
  __shared__ float dg[16];
  __shared__ float red[256];
  int n0 = blockIdx.x * 16, bh = blockIdx.y;
  int b = bh / HH, hh = bh % HH;
  int tid = threadIdx.x;
  for (int i = 0; i < 4; i++) {
    int idx = tid + i * 256; int r = idx >> 6, d = idx & 63;
    unsigned short v = kb[((size_t)(b * NK + n0 + r)) * INNER + hh * 64 + d];
    Aq[r][d] = v;
    Aq2[(d >> 5) * 512 + fragA_idx(r, d & 31)] = v;
  }
  for (int i = 0; i < 68; i++) {
    int idx = tid + i * 256;
    int d = idx / MP, m = idx % MP;
    Bp2[(((m >> 4) * 2 + (d >> 5)) << 9) + fragB_idx(d & 31, m & 15)] = pjt[idx];
  }
  __syncthreads();
  int lane = tid & 31, wv = tid >> 5, half = lane >> 4, lr = lane & 15;
  for (int t = wv; t < 17; t += 8) {
    v8f acc = {0.f, 0.f, 0.f, 0.f, 0.f, 0.f, 0.f, 0.f};
#pragma unroll
    for (int ch = 0; ch < 2; ch++) {
      v16bf a = ld_frag(Aq2 + ch * 512 + (lane << 4));
      v16bf bf = ld_frag(Bp2 + (((t * 2 + ch) * 32 + lane) << 4));
      acc = __builtin_amdgcn_wmma_f32_16x16x32_bf16(false, a, false, bf, (short)0, acc,
                                                    false, false);
    }
#pragma unroll
    for (int r = 0; r < 8; r++) XD[r + 8 * half][t * 16 + lr] = acc[r];
  }
  __syncthreads();
  if (PASS == 1) {
    float lm = -3.4e38f;
    for (int i = tid; i < 16 * MP; i += 256) {
      int m = i % MP;
      if (m < MF) lm = fmaxf(lm, XD[i / MP][m]);
    }
    red[tid] = lm; __syncthreads();
    for (int o = 128; o > 0; o >>= 1) {
      if (tid < o) red[tid] = fmaxf(red[tid], red[tid + o]);
      __syncthreads();
    }
    if (!tid) atomicMax(&mxk[bh], fenc(red[0]));
  } else {
    if (tid < 16) {
      float s = 0.f;
      for (int d = 0; d < 64; d++) { float v = bf2f(Aq[tid][d]); s += v * v; }
      dg[tid] = s * (0.5f * DN_ * DN_);
    }
    __syncthreads();
    float mx = fdec(mxk[bh]);
    for (int i = tid; i < 16 * MP; i += 256) {
      int r = i / MP, m = i % MP;
      float v = (m < MF) ? RATIO_ * (__expf(XD[r][m] - dg[r] - mx) + FEPS_) : 0.f;
      kp[((size_t)bh * NK + n0 + r) * MP + m] = f2bf(v);
    }
  }
}

__global__ void k_ksum(const unsigned short* __restrict__ kp, float* __restrict__ ksum) {
  int bh = blockIdx.x, tid = threadIdx.x;
  int c1 = tid + 256;
  float s0 = 0.f, s1 = 0.f;
  for (int row = 0; row < NK; row++) {
    size_t base = ((size_t)bh * NK + row) * MP;
    s0 += bf2f(kp[base + tid]);
    if (c1 < MP) s1 += bf2f(kp[base + c1]);
  }
  ksum[bh * MP + tid] = s0;
  if (c1 < MP) ksum[bh * MP + c1] = s1;
}

__global__ void k_dinv(const unsigned short* __restrict__ qp, const float* __restrict__ ksum,
                       float* __restrict__ dinv) {
  int n = blockIdx.x * 256 + threadIdx.x;
  int bh = blockIdx.y;
  size_t base = ((size_t)bh * NQ + n) * MP;
  float s = 0.f;
  for (int m = 0; m < MP; m++) s += bf2f(qp[base + m]) * ksum[bh * MP + m];
  dinv[(size_t)bh * NQ + n] = 1.f / s;
}

__global__ void k_glu(const float* __restrict__ hff, unsigned short* __restrict__ u) {
  size_t i = (size_t)blockIdx.x * 256 + threadIdx.x;
  if (i >= (size_t)MQ * FFH) return;
  size_t r = i >> 8; int j = (int)(i & 255);
  float a = hff[r * FFD + j], g = hff[r * FFD + FFH + j];
  float t = 0.7978845608028654f * (a + 0.044715f * a * a * a);
  float ge = 0.5f * a * (1.f + tanhf(t));
  u[i] = f2bf(ge * g);
}

__global__ void k_out(const float* __restrict__ outf, const float* __restrict__ qe,
                      float* __restrict__ out) {
  size_t i = (size_t)blockIdx.x * 256 + threadIdx.x;
  if (i >= (size_t)BB * CC * NQ) return;
  int t = (int)(i & 4095);
  int c = (int)((i >> 12) & 127);
  int b = (int)(i >> 19);
  size_t src = ((size_t)(b * NQ + t)) * CC + c;
  out[i] = outf[src] + qe[src];
}

// ---------------------------------------------------------------------------
// host
// ---------------------------------------------------------------------------

extern "C" void kernel_launch(void* const* d_in, const int* in_sizes, int n_in,
                              void* d_out, int out_size, void* d_ws, size_t ws_size,
                              hipStream_t stream) {
  (void)in_sizes; (void)n_in; (void)out_size; (void)ws_size;
  const float* x = (const float*)d_in[0];
  const float* qP[13]; const float* cP[13];
  for (int i = 0; i < 13; i++) { qP[i] = (const float*)d_in[1 + i]; cP[i] = (const float*)d_in[14 + i]; }
  const float* g_attn = (const float*)d_in[27];
  const float* wq = (const float*)d_in[28]; const float* bq = (const float*)d_in[29];
  const float* wk = (const float*)d_in[30]; const float* bk = (const float*)d_in[31];
  const float* wv = (const float*)d_in[32]; const float* bv = (const float*)d_in[33];
  const float* wo = (const float*)d_in[34]; const float* bo = (const float*)d_in[35];
  const float* proj = (const float*)d_in[36];
  const float* g_ff = (const float*)d_in[37];
  const float* w1 = (const float*)d_in[38]; const float* b1 = (const float*)d_in[39];
  const float* w2 = (const float*)d_in[40]; const float* b2 = (const float*)d_in[41];

  char* W = (char*)d_ws;
  size_t off = 0;
  auto ALLOC = [&](size_t bytes) -> char* {
    char* p = W + off;
    off = (off + bytes + 255) & ~(size_t)255;
    return p;
  };
  float* e_buf = (float*)ALLOC((size_t)BB * 128 * LL * 4);
  float* h_buf = (float*)ALLOC((size_t)BB * 128 * NQ * 4);
  float* meanb = (float*)ALLOC((size_t)BB * 128 * 4);
  float* sesc  = (float*)ALLOC((size_t)BB * 128 * 4);
  float* qe    = (float*)ALLOC((size_t)MQ * CC * 4);
  float* ce    = (float*)ALLOC((size_t)MK * CC * 4);
  unsigned short* xn   = (unsigned short*)ALLOC((size_t)MQ * CC * 2);
  unsigned short* cebf = (unsigned short*)ALLOC((size_t)MK * CC * 2);
  unsigned short* wqb = (unsigned short*)ALLOC((size_t)CC * INNER * 2);
  unsigned short* wkb = (unsigned short*)ALLOC((size_t)CC * INNER * 2);
  unsigned short* wvb = (unsigned short*)ALLOC((size_t)CC * INNER * 2);
  unsigned short* wob = (unsigned short*)ALLOC((size_t)INNER * CC * 2);
  unsigned short* w1b = (unsigned short*)ALLOC((size_t)CC * FFD * 2);
  unsigned short* w2b = (unsigned short*)ALLOC((size_t)FFH * CC * 2);
  unsigned short* pjt = (unsigned short*)ALLOC((size_t)64 * MP * 2);
  unsigned short* qb  = (unsigned short*)ALLOC((size_t)MQ * INNER * 2);
  unsigned short* kb  = (unsigned short*)ALLOC((size_t)MK * INNER * 2);
  unsigned short* vb  = (unsigned short*)ALLOC((size_t)MK * INNER * 2);
  unsigned short* qp  = (unsigned short*)ALLOC((size_t)BHN * NQ * MP * 2);
  unsigned short* kp  = (unsigned short*)ALLOC((size_t)BHN * NK * MP * 2);
  unsigned* mxk = (unsigned*)ALLOC((size_t)BHN * 4);
  float* ksum = (float*)ALLOC((size_t)BHN * MP * 4);
  unsigned short* kvb = (unsigned short*)ALLOC((size_t)BHN * MP * DHD * 2);
  float* dinv = (float*)ALLOC((size_t)BHN * NQ * 4);
  unsigned short* attnb = (unsigned short*)ALLOC((size_t)MQ * INNER * 2);
  float* ybuf = (float*)ALLOC((size_t)MQ * CC * 4);
  unsigned short* ynbf = (unsigned short*)ALLOC((size_t)MQ * CC * 2);
  float* hff  = (float*)ALLOC((size_t)MQ * FFD * 4);
  unsigned short* ubf = (unsigned short*)ALLOC((size_t)MQ * FFH * 2);
  float* outf = (float*)ALLOC((size_t)MQ * CC * 4);

  // ---- MBConv q branch (stride 2) ----
  k_expand<<<dim3(LL / 64, BB), 256, 0, stream>>>(x, qP[0], qP[1], qP[2], e_buf);
  k_dwconv<<<dim3(NQ / 256, 128, BB), 256, 0, stream>>>(e_buf, qP[3], qP[4], qP[5], h_buf, NQ, 2);
  k_semean<<<BB * 128, 256, 0, stream>>>(h_buf, meanb, NQ);
  k_se<<<BB, 128, 0, stream>>>(meanb, qP[6], qP[7], qP[8], qP[9], sesc);
  k_project<<<dim3(NQ / 64, BB), 256, 0, stream>>>(h_buf, sesc, qP[10], qP[11], qP[12], qe, NQ);
  // ---- MBConv c branch (stride 4) ----
  k_expand<<<dim3(LL / 64, BB), 256, 0, stream>>>(x, cP[0], cP[1], cP[2], e_buf);
  k_dwconv<<<dim3(NK / 256, 128, BB), 256, 0, stream>>>(e_buf, cP[3], cP[4], cP[5], h_buf, NK, 4);
  k_semean<<<BB * 128, 256, 0, stream>>>(h_buf, meanb, NK);
  k_se<<<BB, 128, 0, stream>>>(meanb, cP[6], cP[7], cP[8], cP[9], sesc);
  k_project<<<dim3(NK / 64, BB), 256, 0, stream>>>(h_buf, sesc, cP[10], cP[11], cP[12], ce, NK);

  // ---- weight conversions ----
  auto cvt = [&](const float* s, unsigned short* d, size_t n) {
    k_cvt<<<(unsigned)((n + 255) / 256), 256, 0, stream>>>(s, d, n);
  };
  cvt(wq, wqb, (size_t)CC * INNER);
  cvt(wk, wkb, (size_t)CC * INNER);
  cvt(wv, wvb, (size_t)CC * INNER);
  cvt(wo, wob, (size_t)INNER * CC);
  cvt(w1, w1b, (size_t)CC * FFD);
  cvt(w2, w2b, (size_t)FFH * CC);
  k_projT<<<(64 * MP + 255) / 256, 256, 0, stream>>>(proj, pjt);
  k_zero<<<1, 256, 0, stream>>>(mxk, BHN);

  // ---- attention ----
  k_scale_norm<<<MQ / 256, 256, 0, stream>>>(qe, xn, g_attn, MQ);
  cvt(ce, cebf, (size_t)MK * CC);

  GP gq = {0, 0, CC, 1, 0, 0, INNER, 1, 0, 0, INNER, 0, 0, 0, 0, 0, MQ, INNER, CC, 1};
  k_gemm<EPI_BF16><<<dim3(MQ / 64, INNER / 32, 1), 256, 0, stream>>>(xn, wqb, qb, bq, nullptr, nullptr, gq);
  GP gk = {0, 0, CC, 1, 0, 0, INNER, 1, 0, 0, INNER, 0, 0, 0, 0, 0, MK, INNER, CC, 1};
  k_gemm<EPI_BF16><<<dim3(MK / 64, INNER / 32, 1), 256, 0, stream>>>(cebf, wkb, kb, bk, nullptr, nullptr, gk);
  k_gemm<EPI_BF16><<<dim3(MK / 64, INNER / 32, 1), 256, 0, stream>>>(cebf, wvb, vb, bv, nullptr, nullptr, gk);

  k_qp<<<dim3(NQ / 16, BHN), 256, 0, stream>>>(qb, pjt, qp);
  k_kxd<1><<<dim3(NK / 16, BHN), 256, 0, stream>>>(kb, pjt, mxk, nullptr);
  k_kxd<2><<<dim3(NK / 16, BHN), 256, 0, stream>>>(kb, pjt, mxk, kp);
  k_ksum<<<BHN, 256, 0, stream>>>(kp, ksum);

  // kv[bh][m][d] = sum_k kp[bh][k][m] * v[b][k][h*64+d]
  GP gkv = {(long)HH * NK * MP, (long)NK * MP, 1, MP,
            (long)NK * INNER, 64, INNER, 1,
            (long)HH * MP * DHD, (long)MP * DHD, DHD,
            0, 0, 0, 0, 0, MP, DHD, NK, HH};
  k_gemm<EPI_BF16><<<dim3((MP + 63) / 64, DHD / 32, BHN), 256, 0, stream>>>(kp, vb, kvb, nullptr, nullptr, nullptr, gkv);

  k_dinv<<<dim3(NQ / 256, BHN), 256, 0, stream>>>(qp, ksum, dinv);

  // attn[b][n][h*64+d] = dinv * sum_m qp[bh][n][m] * kv[bh][m][d]
  GP gat = {(long)HH * NQ * MP, (long)NQ * MP, MP, 1,
            (long)HH * MP * DHD, (long)MP * DHD, DHD, 1,
            (long)NQ * INNER, 64, INNER,
            0, 0, 0, (long)HH * NQ, (long)NQ, NQ, DHD, MP, HH};
  k_gemm<EPI_BF16_RS><<<dim3(NQ / 64, DHD / 32, BHN), 256, 0, stream>>>(qp, kvb, attnb, nullptr, nullptr, dinv, gat);

  // y = qe + attn @ wo + bo
  GP gwo = {0, 0, INNER, 1, 0, 0, CC, 1, 0, 0, CC, 0, 0, CC, 0, 0, MQ, CC, INNER, 1};
  k_gemm<EPI_F32_RES><<<dim3(MQ / 64, CC / 32, 1), 256, 0, stream>>>(attnb, wob, ybuf, bo, qe, nullptr, gwo);

  // ---- GLU feed-forward ----
  k_scale_norm<<<MQ / 256, 256, 0, stream>>>(ybuf, ynbf, g_ff, MQ);
  GP gw1 = {0, 0, CC, 1, 0, 0, FFD, 1, 0, 0, FFD, 0, 0, 0, 0, 0, MQ, FFD, CC, 1};
  k_gemm<EPI_F32><<<dim3(MQ / 64, FFD / 32, 1), 256, 0, stream>>>(ynbf, w1b, hff, b1, nullptr, nullptr, gw1);
  k_glu<<<(unsigned)(((size_t)MQ * FFH + 255) / 256), 256, 0, stream>>>(hff, ubf);
  GP gw2 = {0, 0, FFH, 1, 0, 0, CC, 1, 0, 0, CC, 0, 0, CC, 0, 0, MQ, CC, FFH, 1};
  k_gemm<EPI_F32_RES><<<dim3(MQ / 64, CC / 32, 1), 256, 0, stream>>>(ubf, w2b, outf, b2, ybuf, nullptr, gw2);

  // ---- outer residual + transpose to [B][C][T] ----
  k_out<<<(unsigned)(((size_t)BB * CC * NQ + 255) / 256), 256, 0, stream>>>(outf, qe, (float*)d_out);
}
